// Attention_49392123904386
// MI455X (gfx1250) — matrix operand order repaired and checked by reading, not compiled, not run on hardware
//
#include <hip/hip_runtime.h>

// ---------------------------------------------------------------------------
// SAGAN self-attention for MI455X (gfx1250, wave32, WMMA 16x16x32 f16)
//   x:[8,4096,256] f32 -> f16 projections -> flash attention -> f32 out
//   V tiles staged LDS-side by the Tensor Data Mover (TENSORcnt), ping-pong.
// ---------------------------------------------------------------------------

typedef __attribute__((ext_vector_type(16))) _Float16 v16h;
typedef __attribute__((ext_vector_type(8)))  _Float16 v8h;
typedef __attribute__((ext_vector_type(4)))  _Float16 v4h;
typedef __attribute__((ext_vector_type(8)))  float    v8f;
typedef __attribute__((ext_vector_type(4)))  float    v4f;
typedef __attribute__((ext_vector_type(4)))  unsigned int v4u;
typedef __attribute__((ext_vector_type(8)))  int      v8i;
typedef __attribute__((ext_vector_type(4)))  int      v4i;

union Frag16 { v16h v; v8h h[2]; };

#define WMMA_F16(a, b, c) \
  __builtin_amdgcn_wmma_f32_16x16x32_f16(false, (a), false, (b), (short)0, (c), false, false)

#define NB   8
#define NN   4096
#define NC   256
#define NCQ  32
#define BM   128   // query rows per workgroup
#define KB   64    // key block
#define VSTRIDE 72 // LDS half-stride per V column (144B, 16B aligned, 36-dword banks)

#if __has_builtin(__builtin_amdgcn_tensor_load_to_lds) && \
    __has_builtin(__builtin_amdgcn_s_wait_tensorcnt)
#define USE_TDM 1
#endif

// ---------------------------------------------------------------------------
// Kernel 1: transpose + convert weights to f16 (WxT[n][k] = Wx[k][n])
// ---------------------------------------------------------------------------
__global__ void prep_weights_kernel(const float* __restrict__ Wf,
                                    const float* __restrict__ Wg,
                                    const float* __restrict__ Wh,
                                    _Float16* __restrict__ WfT,
                                    _Float16* __restrict__ WgT,
                                    _Float16* __restrict__ WhT) {
  const int t = threadIdx.x;  // 0..255
  for (int k = 0; k < NC; ++k)
    WhT[t * NC + k] = (_Float16)Wh[k * NC + t];
  if (t < NCQ) {
    for (int k = 0; k < NC; ++k)
      WfT[t * NC + k] = (_Float16)Wf[k * NCQ + t];
  } else if (t < 2 * NCQ) {
    const int n = t - NCQ;
    for (int k = 0; k < NC; ++k)
      WgT[n * NC + k] = (_Float16)Wg[k * NCQ + n];
  }
}

// ---------------------------------------------------------------------------
// Kernel 2: projections f,g (row-major f16) and h (transposed f16 hT[col][row])
//   grid = 256 WGs x 256 threads (8 waves), 128 rows per WG, 16 rows per wave
// ---------------------------------------------------------------------------
__global__ void __launch_bounds__(256) proj_kernel(
    const float*    __restrict__ x,
    const _Float16* __restrict__ WfT,
    const _Float16* __restrict__ WgT,
    const _Float16* __restrict__ WhT,
    const float*    __restrict__ bf,
    const float*    __restrict__ bg,
    const float*    __restrict__ bh,
    _Float16* __restrict__ fK,
    _Float16* __restrict__ gQ,
    _Float16* __restrict__ hT) {
  __shared__ __align__(16) _Float16 smem[128 * 136];  // hT staging, 2 col-halves

  const int t     = threadIdx.x;
  const int rBase = blockIdx.x * BM;
  const int w     = t >> 5;
  const int L     = t & 31;
  const int l16   = L & 15;
  const int hw    = L >> 4;

  v8f accH[16], accF[2], accG[2];
  const v8f vz = {0.f, 0.f, 0.f, 0.f, 0.f, 0.f, 0.f, 0.f};
#pragma unroll
  for (int i = 0; i < 16; ++i) accH[i] = vz;
#pragma unroll
  for (int i = 0; i < 2; ++i) { accF[i] = vz; accG[i] = vz; }

  const int myRow = rBase + w * 16 + l16;  // A-fragment row for this lane

  for (int kk = 0; kk < 8; ++kk) {
    // A fragment (16x32 f16) built from global f32 x.
    // Layout: halves[0..7] = k kOff..kOff+7 ; halves[8..15] = k kOff+16..kOff+23
    const int kOffA = kk * 32 + hw * 8;
    const float* xr = x + (size_t)myRow * NC + kOffA;
    v4f x0 = *(const v4f*)(xr + 0);
    v4f x1 = *(const v4f*)(xr + 4);
    v4f x2 = *(const v4f*)(xr + 16);
    v4f x3 = *(const v4f*)(xr + 20);
    Frag16 a;
    v8h a0, a1;
    a0.s0 = (_Float16)x0.x; a0.s1 = (_Float16)x0.y; a0.s2 = (_Float16)x0.z; a0.s3 = (_Float16)x0.w;
    a0.s4 = (_Float16)x1.x; a0.s5 = (_Float16)x1.y; a0.s6 = (_Float16)x1.z; a0.s7 = (_Float16)x1.w;
    a1.s0 = (_Float16)x2.x; a1.s1 = (_Float16)x2.y; a1.s2 = (_Float16)x2.z; a1.s3 = (_Float16)x2.w;
    a1.s4 = (_Float16)x3.x; a1.s5 = (_Float16)x3.y; a1.s6 = (_Float16)x3.z; a1.s7 = (_Float16)x3.w;
    a.h[0] = a0; a.h[1] = a1;

    const int kOffB = kk * 32 + hw * 8;  // B layout: k kOff..kOff+7, kOff+16..kOff+23 per lane
#pragma unroll
    for (int tt = 0; tt < 16; ++tt) {
      Frag16 bfr;
      const _Float16* bp = WhT + (size_t)(tt * 16 + l16) * NC + kOffB;
      bfr.h[0] = *(const v8h*)bp;
      bfr.h[1] = *(const v8h*)(bp + 16);
      accH[tt] = WMMA_F16(a.v, bfr.v, accH[tt]);
    }
#pragma unroll
    for (int tt = 0; tt < 2; ++tt) {
      Frag16 bfrF, bfrG;
      const _Float16* fp = WfT + (size_t)(tt * 16 + l16) * NC + kOffB;
      const _Float16* gp = WgT + (size_t)(tt * 16 + l16) * NC + kOffB;
      bfrF.h[0] = *(const v8h*)fp; bfrF.h[1] = *(const v8h*)(fp + 16);
      bfrG.h[0] = *(const v8h*)gp; bfrG.h[1] = *(const v8h*)(gp + 16);
      accF[tt] = WMMA_F16(a.v, bfrF.v, accF[tt]);
      accG[tt] = WMMA_F16(a.v, bfrG.v, accG[tt]);
    }
  }

  // f/g: add bias, store row-major f16. C/D layout: row = j + 8*hw, col = tt*16+l16.
#pragma unroll
  for (int tt = 0; tt < 2; ++tt) {
    const int n = tt * 16 + l16;
    const float bfv = bf[n];
    const float bgv = bg[n];
#pragma unroll
    for (int j = 0; j < 8; ++j) {
      const int m = rBase + w * 16 + hw * 8 + j;
      fK[(size_t)m * NCQ + n] = (_Float16)(accF[tt][j] + bfv);
      gQ[(size_t)m * NCQ + n] = (_Float16)(accG[tt][j] + bgv);
    }
  }

  // h: add bias, transpose through LDS (two 128-column halves), store hT[col][row].
  const int bIdx = rBase >> 12;   // batch
  const int rIn  = rBase & 4095;  // row within batch
  for (int r = 0; r < 2; ++r) {
#pragma unroll
    for (int tt = 0; tt < 8; ++tt) {
      const int ttg = r * 8 + tt;
      const int n   = ttg * 16 + l16;
      const float bhv = bh[n];
      const int colL = tt * 16 + l16;      // column within this half
      const int rloc = w * 16 + hw * 8;    // local row of j=0
      v4h p;
      p.x = (_Float16)(accH[ttg][0] + bhv);
      p.y = (_Float16)(accH[ttg][1] + bhv);
      p.z = (_Float16)(accH[ttg][2] + bhv);
      p.w = (_Float16)(accH[ttg][3] + bhv);
      *(v4h*)&smem[colL * 136 + rloc] = p;
      p.x = (_Float16)(accH[ttg][4] + bhv);
      p.y = (_Float16)(accH[ttg][5] + bhv);
      p.z = (_Float16)(accH[ttg][6] + bhv);
      p.w = (_Float16)(accH[ttg][7] + bhv);
      *(v4h*)&smem[colL * 136 + rloc + 4] = p;
    }
    __syncthreads();
    {
      const int col = t & 127;
      const int rh  = (t >> 7) * 64;  // rows 0..63 or 64..127
      _Float16* dst = hT + ((size_t)(bIdx * NC + r * 128 + col)) * NN + rIn + rh;
#pragma unroll
      for (int i = 0; i < 8; ++i)
        *(v8h*)(dst + i * 8) = *(const v8h*)&smem[col * 136 + rh + i * 8];
    }
    __syncthreads();
  }
}

// ---------------------------------------------------------------------------
// TDM: stage one V tile [256 cols x 64 keys, f16] from hT into LDS with the
// Tensor Data Mover.  2D tile: dim0 = 64 contiguous halves (one column's key
// block), dim1 = 256 rows at stride 4096.  pad_enable inserts 4 DWORDs after
// every 32 DWORDs (=128B row) -> LDS row stride 144B = VSTRIDE halves.
// ---------------------------------------------------------------------------
#ifdef USE_TDM
__device__ __forceinline__ void tdm_stage_v(const _Float16* gsrc, _Float16* ldst) {
  const unsigned long long ga = (unsigned long long)gsrc;
  const unsigned int laddr = (unsigned int)(unsigned long long)ldst;  // LDS byte offset
  v4u g0;
  g0.x = 1u;                                       // count=1 (valid user D#)
  g0.y = laddr;                                    // lds_addr
  g0.z = (unsigned int)ga;                         // global_addr[31:0]
  g0.w = (unsigned int)(ga >> 32) | (2u << 30);    // global_addr[56:32] | type=2
  v8i g1;
  g1.s0 = (int)((1u << 16)        // data_size = 2 bytes
              | (1u << 20)        // pad_enable
              | (4u << 22)        // pad_interval: every 32 DWORDs (128B)
              | (3u << 25));      // pad_amount: 4 DWORDs (16B)
  g1.s1 = (int)(4096u << 16);     // tensor_dim0 = 4096 (low16 at [31:16])
  g1.s2 = (int)(256u << 16);      // tensor_dim0 hi=0 | tensor_dim1 = 256
  g1.s3 = (int)(64u << 16);       // tensor_dim1 hi=0 | tile_dim0 = 64
  g1.s4 = (int)256;               // tile_dim1 = 256 | tile_dim2 = 0
  g1.s5 = (int)4096;              // tensor_dim0_stride[31:0] = 4096
  g1.s6 = 0;                      // stride hi | tensor_dim1_stride lo
  g1.s7 = 0;
  const v4i gz = {0, 0, 0, 0};
#if __clang_major__ >= 23
  const v8i gz8 = {0, 0, 0, 0, 0, 0, 0, 0};
  __builtin_amdgcn_tensor_load_to_lds(g0, g1, gz, gz, gz8, 0);
#else
  __builtin_amdgcn_tensor_load_to_lds(g0, g1, gz, gz, 0);
#endif
}
#endif

// ---------------------------------------------------------------------------
// Kernel 3: flash attention.  Q=g, K=f, V=h (via hT).
//   grid = 8 batches * 32 q-blocks; 8 waves, 16 query rows per wave, d_v=256.
//   Dynamic LDS: VT ping [0,36864)B, VT pong [36864,73728)B, P [73728,92160)B
// ---------------------------------------------------------------------------
__global__ void __launch_bounds__(256) attn_kernel(
    const float*    __restrict__ x,
    const _Float16* __restrict__ gQ,
    const _Float16* __restrict__ fK,
    const _Float16* __restrict__ hT,
    const float*    __restrict__ gamma,
    float* __restrict__ out) {
  extern __shared__ _Float16 dynsmem[];

  const int t     = threadIdx.x;
  const int b     = blockIdx.x >> 5;
  const int qb    = blockIdx.x & 31;
  const int qBase = qb * BM;
  const int w     = t >> 5;
  const int L     = t & 31;
  const int l16   = L & 15;
  const int hw    = L >> 4;

  const _Float16* hTb = hT + (size_t)b * NC * NN;
  const _Float16* fKb = fK + (size_t)b * NN * NCQ;
  const _Float16* gQb = gQ + (size_t)b * NN * NCQ;

  _Float16* VT[2];
  VT[0] = dynsmem;                       // [256][VSTRIDE]
  VT[1] = dynsmem + NC * VSTRIDE;        // [256][VSTRIDE]
  _Float16* ldsP = dynsmem + 2 * NC * VSTRIDE + w * (16 * VSTRIDE);  // [16][VSTRIDE]

  // Q fragment (A 16x32): row = qBase + w*16 + l16, chunks at k hw*8 and hw*8+16
  Frag16 qf;
  {
    const _Float16* qp = gQb + (size_t)(qBase + w * 16 + l16) * NCQ + hw * 8;
    qf.h[0] = *(const v8h*)qp;
    qf.h[1] = *(const v8h*)(qp + 16);
  }

  const v8f vz = {0.f, 0.f, 0.f, 0.f, 0.f, 0.f, 0.f, 0.f};
  v8f accO[16];
#pragma unroll
  for (int i = 0; i < 16; ++i) accO[i] = vz;
  float rowMax[8], rowSum[8];
#pragma unroll
  for (int j = 0; j < 8; ++j) { rowMax[j] = -3.0e38f; rowSum[j] = 0.f; }

  // ---- prologue: stage key block 0 into VT[0] ----
#ifdef USE_TDM
  if (w == 0) {
    tdm_stage_v(hTb, VT[0]);
    __builtin_amdgcn_s_wait_tensorcnt(0);
  }
#else
  {
    const _Float16* src = hTb + (size_t)t * NN;
#pragma unroll
    for (int i = 0; i < 8; ++i)
      *(v8h*)&VT[0][t * VSTRIDE + i * 8] = *(const v8h*)(src + i * 8);
  }
#endif
  __syncthreads();

#pragma unroll 1
  for (int kb = 0; kb < NN / KB; ++kb) {
    _Float16* cur = VT[kb & 1];
    _Float16* nxt = VT[(kb + 1) & 1];

    // ---- kick off async stage of the NEXT key block (overlaps compute) ----
    if (kb + 1 < NN / KB) {
#ifdef USE_TDM
      if (w == 0) tdm_stage_v(hTb + (size_t)(kb + 1) * KB, nxt);
#else
      const _Float16* src = hTb + (size_t)t * NN + (kb + 1) * KB;
      __builtin_prefetch(src, 0, 1);
#pragma unroll
      for (int i = 0; i < 8; ++i)
        *(v8h*)&nxt[t * VSTRIDE + i * 8] = *(const v8h*)(src + i * 8);
#endif
    }

    // ---- S = Q * K^T : 4 WMMAs (16 queries x 64 keys) ----
    v8f sF[4];
#pragma unroll
    for (int tt = 0; tt < 4; ++tt) {
      Frag16 kf;
      const _Float16* kp = fKb + (size_t)(kb * KB + tt * 16 + l16) * NCQ + hw * 8;
      kf.h[0] = *(const v8h*)kp;
      kf.h[1] = *(const v8h*)(kp + 16);
      sF[tt] = WMMA_F16(qf.v, kf.v, vz);
    }

    // ---- online softmax (row = j + 8*hw; cols across 16 lanes + 4 tiles) ----
    float corr[8];
#pragma unroll
    for (int j = 0; j < 8; ++j) {
      float mt = fmaxf(fmaxf(sF[0][j], sF[1][j]), fmaxf(sF[2][j], sF[3][j]));
      mt = fmaxf(mt, __shfl_xor(mt, 1, 32));
      mt = fmaxf(mt, __shfl_xor(mt, 2, 32));
      mt = fmaxf(mt, __shfl_xor(mt, 4, 32));
      mt = fmaxf(mt, __shfl_xor(mt, 8, 32));
      const float mNew = fmaxf(rowMax[j], mt);
      const float c = __expf(rowMax[j] - mNew);
      corr[j] = c;
      float ps = 0.f;
#pragma unroll
      for (int tt = 0; tt < 4; ++tt) {
        const float p = __expf(sF[tt][j] - mNew);
        sF[tt][j] = p;
        ps += p;
      }
      ps += __shfl_xor(ps, 1, 32);
      ps += __shfl_xor(ps, 2, 32);
      ps += __shfl_xor(ps, 4, 32);
      ps += __shfl_xor(ps, 8, 32);
      rowSum[j] = rowSum[j] * c + ps;
      rowMax[j] = mNew;
    }
#pragma unroll
    for (int tt = 0; tt < 16; ++tt) {
#pragma unroll
      for (int j = 0; j < 8; ++j) accO[tt][j] *= corr[j];
    }

    // ---- P (C/D layout) -> LDS row-major f16, then reload as A fragments ----
#pragma unroll
    for (int tt = 0; tt < 4; ++tt) {
#pragma unroll
      for (int j = 0; j < 8; ++j)
        ldsP[(j + hw * 8) * VSTRIDE + tt * 16 + l16] = (_Float16)sF[tt][j];
    }
    asm volatile("" ::: "memory");  // keep LDS store->load order (same-wave DS is in-order)

    // ---- O += P * V : 2 k-halves x 16 col tiles = 32 WMMAs ----
#pragma unroll
    for (int kf2 = 0; kf2 < 2; ++kf2) {
      Frag16 pf;
      const _Float16* pp = &ldsP[l16 * VSTRIDE + kf2 * 32 + hw * 8];
      pf.h[0] = *(const v8h*)pp;
      pf.h[1] = *(const v8h*)(pp + 16);
#pragma unroll
      for (int tt = 0; tt < 16; ++tt) {
        Frag16 vf;
        const _Float16* vp = &cur[(tt * 16 + l16) * VSTRIDE + kf2 * 32 + hw * 8];
        vf.h[0] = *(const v8h*)vp;
        vf.h[1] = *(const v8h*)(vp + 16);
        accO[tt] = WMMA_F16(pf.v, vf.v, accO[tt]);
      }
    }

    // ---- make next tile visible: TDM done (wave 0) then block barrier ----
#ifdef USE_TDM
    if (w == 0) __builtin_amdgcn_s_wait_tensorcnt(0);
#endif
    __syncthreads();
  }

  // ---- epilogue: out = gamma * (O / l) + x ----
  const float g0 = gamma[0];
  float inv[8];
#pragma unroll
  for (int j = 0; j < 8; ++j) inv[j] = 1.0f / rowSum[j];
#pragma unroll
  for (int tt = 0; tt < 16; ++tt) {
#pragma unroll
    for (int j = 0; j < 8; ++j) {
      const int m = qBase + w * 16 + hw * 8 + j;
      const size_t idx = ((size_t)(b * NN + m)) * NC + tt * 16 + l16;
      out[idx] = g0 * (accO[tt][j] * inv[j]) + x[idx];
    }
  }
}

// ---------------------------------------------------------------------------
// Host launch
// ---------------------------------------------------------------------------
extern "C" void kernel_launch(void* const* d_in, const int* in_sizes, int n_in,
                              void* d_out, int out_size, void* d_ws, size_t ws_size,
                              hipStream_t stream) {
  (void)in_sizes; (void)n_in; (void)out_size; (void)ws_size;
  const float* x   = (const float*)d_in[0];
  const float* Wf  = (const float*)d_in[1];
  const float* Wg  = (const float*)d_in[2];
  const float* Wh  = (const float*)d_in[3];
  const float* bf  = (const float*)d_in[4];
  const float* bg  = (const float*)d_in[5];
  const float* bh  = (const float*)d_in[6];
  const float* gam = (const float*)d_in[7];
  float* out = (float*)d_out;

  char* ws = (char*)d_ws;
  _Float16* WfT = (_Float16*)(ws + 0);                       //  16 KB
  _Float16* WgT = (_Float16*)(ws + 16384);                   //  16 KB
  _Float16* WhT = (_Float16*)(ws + 32768);                   // 128 KB
  _Float16* fK  = (_Float16*)(ws + 163840);                  //   2 MB
  _Float16* gQ  = (_Float16*)(ws + 163840 + 2097152);        //   2 MB
  _Float16* hT  = (_Float16*)(ws + 163840 + 2 * 2097152);    //  16 MB

  // dynamic LDS for attn: 2 x (256*72) + 8 x (16*72) halves = 92160 bytes
  const size_t attn_lds = (size_t)(2 * NC * VSTRIDE + 8 * 16 * VSTRIDE) * sizeof(_Float16);

  prep_weights_kernel<<<1, 256, 0, stream>>>(Wf, Wg, Wh, WfT, WgT, WhT);
  proj_kernel<<<256, 256, 0, stream>>>(x, WfT, WgT, WhT, bf, bg, bh, fK, gQ, hT);
  attn_kernel<<<NB * (NN / BM), 256, attn_lds, stream>>>(x, gQ, fK, hT, gam, out);
}
